// AutoCorrelation_82875688944218
// MI455X (gfx1250) — matrix operand order repaired and checked by reading, not compiled, not run on hardware
//
#include <hip/hip_runtime.h>
#include <hip/hip_bf16.h>
#include <math.h>

// ---------------------------------------------------------------------------
// Problem constants (fixed by the reference harness)
// ---------------------------------------------------------------------------
#define BB     8
#define LL     4096
#define DD     512
#define HH     8
#define DH     64
#define TOPK   16          // int(2 * ln(4096)) == 16
#define FFT_N  4096
#define FFT_LOG2 12
#define NEG_INF (-3.402823466e38f)

typedef __bf16  bf16_t;
typedef __bf16  v16bf __attribute__((ext_vector_type(16)));
typedef float   v8f   __attribute__((ext_vector_type(8)));
typedef unsigned int v4u __attribute__((ext_vector_type(4)));
typedef int     v4i   __attribute__((ext_vector_type(4)));
typedef int     v8i   __attribute__((ext_vector_type(8)));

// ---------------------------------------------------------------------------
// Kernel 0: one-time fp32 -> bf16 weight conversion (hardware cvt path).
// ---------------------------------------------------------------------------
__global__ __launch_bounds__(256)
void ac_w_convert(const float* __restrict__ W, bf16_t* __restrict__ Wb) {
    const int i = (blockIdx.x * 256 + threadIdx.x) * 4;
    const float4 f = *reinterpret_cast<const float4*>(W + i);
    bf16_t* d = Wb + i;
    d[0] = (bf16_t)f.x; d[1] = (bf16_t)f.y; d[2] = (bf16_t)f.z; d[3] = (bf16_t)f.w;
}

// ---------------------------------------------------------------------------
// Kernel 1: projection GEMM  Y[m,n] = sum_k X[m,k] * W[n,k] + bias[n]
// bf16 WMMA (v_wmma_f32_16x16x32_bf16), fp32 accumulate.
// Block = 256 threads (8 waves); block tile 256(M) x 64(N); wave tile 32x64:
// two A fragments share each B fragment -> 2 independent WMMAs per ds_load,
// hiding LDS latency and halving B traffic per FLOP.
// Weight slab (64 rows x 512 K, bf16, 64 KB) DMA'd into LDS by the TDM.
// ---------------------------------------------------------------------------
__device__ __forceinline__ v16bf ac_load_a(const float* xrow, int ka) {
    const float4 fa0 = *reinterpret_cast<const float4*>(xrow + ka);
    const float4 fa1 = *reinterpret_cast<const float4*>(xrow + ka + 4);
    const float4 fa2 = *reinterpret_cast<const float4*>(xrow + ka + 16);
    const float4 fa3 = *reinterpret_cast<const float4*>(xrow + ka + 20);
    v16bf a;                                  // hardware v_cvt_pk_bf16_f32 path
    a[0]=(bf16_t)fa0.x; a[1]=(bf16_t)fa0.y; a[2]=(bf16_t)fa0.z; a[3]=(bf16_t)fa0.w;
    a[4]=(bf16_t)fa1.x; a[5]=(bf16_t)fa1.y; a[6]=(bf16_t)fa1.z; a[7]=(bf16_t)fa1.w;
    a[8]=(bf16_t)fa2.x; a[9]=(bf16_t)fa2.y; a[10]=(bf16_t)fa2.z; a[11]=(bf16_t)fa2.w;
    a[12]=(bf16_t)fa3.x; a[13]=(bf16_t)fa3.y; a[14]=(bf16_t)fa3.z; a[15]=(bf16_t)fa3.w;
    return a;
}

__global__ __launch_bounds__(256)
void ac_proj_gemm(const float* __restrict__ X, const bf16_t* __restrict__ Wb,
                  const float* __restrict__ bias, float* __restrict__ Y) {
    __shared__ bf16_t Wl[64 * DD];                 // 64 KB

    const int tid  = threadIdx.x;
    const int lane = tid & 31;
    const int wave = tid >> 5;
    const int n0   = blockIdx.x * 64;
    const int m0   = blockIdx.y * 256 + wave * 32;

    // ---- TDM: DMA the 64x512 bf16 weight tile (row-major, contiguous) into LDS.
    if (wave == 0) {
        const unsigned long long ga =
            (unsigned long long)(uintptr_t)(Wb + (size_t)n0 * DD);   // tile start
        const unsigned ldsAddr = (unsigned)(uintptr_t)(&Wl[0]);
        // D# group 0: count=1, lds_addr, global_addr[56:0], type=2
        v4u g0 = { 1u,
                   ldsAddr,
                   (unsigned)(ga & 0xffffffffu),
                   (unsigned)((ga >> 32) & 0x01ffffffu) | (2u << 30) };
        // D# group 1: wg_mask=0, data_size=1(2B); tensor_dim0=512, tensor_dim1=512;
        // tile_dim0=512, tile_dim1=64, tile_dim2=0; tensor_dim0_stride=512.
        v8i g1 = { (int)(1u << 16),          // data_size=2B
                   (int)(512u << 16),        // tensor_dim0 lo16 << 16
                   (int)(512u << 16),        // tensor_dim0 hi16 | tensor_dim1 lo16<<16
                   (int)(512u << 16),        // tensor_dim1 hi16 | tile_dim0<<16
                   64,                       // tile_dim1 | tile_dim2<<16
                   512,                      // tensor_dim0_stride lo32
                   0,                        // stride0 hi16 | stride1 lo16<<16
                   0 };                      // tensor_dim1_stride hi32
        v4i gz = { 0, 0, 0, 0 };
#if __clang_major__ >= 23
        v8i gz8 = { 0, 0, 0, 0, 0, 0, 0, 0 };
        __builtin_amdgcn_tensor_load_to_lds(g0, g1, gz, gz, gz8, 0);
#else
        __builtin_amdgcn_tensor_load_to_lds(g0, g1, gz, gz, 0);
#endif
    }
    __builtin_amdgcn_s_wait_tensorcnt(0);
    __syncthreads();

    v8f acc[8];
    #pragma unroll
    for (int j = 0; j < 8; ++j) acc[j] = (v8f)(0.0f);

    const bool  hiHalf = (lane & 16) != 0;
    const float* xrow0 = X + (size_t)(m0 + (lane & 15)) * DD;
    const float* xrow1 = X + (size_t)(m0 + 16 + (lane & 15)) * DD;

    for (int k0 = 0; k0 < DD; k0 += 32) {
        __builtin_prefetch(xrow0 + k0 + 32, 0, 1);   // global_prefetch (speculative)
        __builtin_prefetch(xrow1 + k0 + 32, 0, 1);

        // --- two A fragments: lane m holds K {0..7, 16..23} (+8 on hi lanes) ---
        const int ka = k0 + (hiHalf ? 8 : 0);
        const v16bf a0 = ac_load_a(xrow0, ka);
        const v16bf a1 = ac_load_a(xrow1, ka);

        // --- 4 B fragments from LDS; each feeds 2 independent WMMAs ---
        const int kb = k0 + (hiHalf ? 8 : 0);
        #pragma unroll
        for (int j = 0; j < 4; ++j) {
            const int nrow = j * 16 + (lane & 15);
            const bf16_t* bp = &Wl[nrow * DD + kb];
            v16bf b;
            #pragma unroll
            for (int e = 0; e < 16; ++e) b[e] = bp[e < 8 ? e : e + 8];
            acc[j] = __builtin_amdgcn_wmma_f32_16x16x32_bf16(
                false, a0, false, b, (short)0, acc[j], false, false);
            acc[4 + j] = __builtin_amdgcn_wmma_f32_16x16x32_bf16(
                false, a1, false, b, (short)0, acc[4 + j], false, false);
        }
    }

    // Epilogue: C/D layout — lane = N, VGPR r = M row (hi lanes: M+8).
    const int mloc = hiHalf ? 8 : 0;
    #pragma unroll
    for (int j = 0; j < 4; ++j) {
        const int col = n0 + j * 16 + (lane & 15);
        const float bv = bias[col];
        #pragma unroll
        for (int r = 0; r < 8; ++r) {
            Y[(size_t)(m0 + mloc + r) * DD + col]      = acc[j][r] + bv;
            Y[(size_t)(m0 + 16 + mloc + r) * DD + col] = acc[4 + j][r] + bv;
        }
    }
}

// ---------------------------------------------------------------------------
// Kernel 2: FFT autocorrelation. One block per (b, h, c) channel.
// z = q + i*k  -> FFT -> split spectra -> P = Qf*conj(Kf) -> IFFT -> corr.
// ---------------------------------------------------------------------------
__device__ __forceinline__ void fft_stages_lds(float2* Z, int tid, float sign) {
    // bit-reverse permutation
    for (int j = tid; j < FFT_N; j += 256) {
        const int r = __brev((unsigned)j) >> (32 - FFT_LOG2);
        if (r > j) { float2 t = Z[j]; Z[j] = Z[r]; Z[r] = t; }
    }
    __syncthreads();
    // iterative DIT radix-2
    for (int s = 1; s <= FFT_LOG2; ++s) {
        const int m = 1 << s, half = m >> 1;
        for (int i = tid; i < FFT_N / 2; i += 256) {
            const int grp = i >> (s - 1);
            const int pos = i & (half - 1);
            const int i0 = grp * m + pos, i1 = i0 + half;
            const float ang = sign * -6.283185307179586f * (float)pos / (float)m;
            float sn, cs; __sincosf(ang, &sn, &cs);
            const float2 zb = Z[i1];
            const float tr = zb.x * cs - zb.y * sn;
            const float ti = zb.x * sn + zb.y * cs;
            const float2 za = Z[i0];
            Z[i0] = make_float2(za.x + tr, za.y + ti);
            Z[i1] = make_float2(za.x - tr, za.y - ti);
        }
        __syncthreads();
    }
}

__global__ __launch_bounds__(256)
void ac_fft_corr(const float* __restrict__ q, const float* __restrict__ k,
                 float* __restrict__ corr) {
    __shared__ float2 Z[FFT_N];                      // 32 KB
    const int chan = blockIdx.x;                     // = b*512 + h*64 + c
    const int b    = chan >> 9;
    const int col  = chan & 511;
    const size_t base = (size_t)b * LL * DD + col;
    const int tid = threadIdx.x;

    for (int t = tid; t < LL; t += 256)
        Z[t] = make_float2(q[base + (size_t)t * DD], k[base + (size_t)t * DD]);
    __syncthreads();

    fft_stages_lds(Z, tid, 1.0f);                    // forward

    // Hermitian split + cross-spectrum, done read-all / barrier / write-all.
    float Pr[16], Pi[16];
    const int j0 = tid * 16;
    #pragma unroll
    for (int i = 0; i < 16; ++i) {
        const int j = j0 + i;
        const float2 zj = Z[j];
        const float2 zk = Z[(FFT_N - j) & (FFT_N - 1)];
        const float Qr = 0.5f * (zj.x + zk.x), Qi = 0.5f * (zj.y - zk.y);
        const float Kr = 0.5f * (zj.y + zk.y), Ki = 0.5f * (zk.x - zj.x);
        Pr[i] = Qr * Kr + Qi * Ki;                   // Qf * conj(Kf)
        Pi[i] = Qi * Kr - Qr * Ki;
    }
    __syncthreads();
    #pragma unroll
    for (int i = 0; i < 16; ++i) Z[j0 + i] = make_float2(Pr[i], Pi[i]);
    __syncthreads();

    fft_stages_lds(Z, tid, -1.0f);                   // inverse (unscaled)

    const float inv = 1.0f / (float)FFT_N;
    float* crow = corr + (size_t)chan * LL;
    for (int t = tid; t < LL; t += 256) crow[t] = Z[t].x * inv;
}

// ---------------------------------------------------------------------------
// Kernel 3: top-16 over tau per channel row + softmax of the 16 scores.
// Row staged into LDS via async global->LDS copies (ASYNCcnt path).
// ---------------------------------------------------------------------------
__global__ __launch_bounds__(256)
void ac_topk_softmax(const float* __restrict__ corr, float* __restrict__ wOut,
                     int* __restrict__ iOut) {
    __shared__ float vals[LL];                       // 16 KB
    __shared__ float redv[256];
    __shared__ int   redi[256];
    __shared__ float tv[TOPK];
    __shared__ int   ti[TOPK];

    const int row = blockIdx.x;
    const int tid = threadIdx.x;
    const float* crow = corr + (size_t)row * LL;

    // async copy: each lane moves 16B to its own LDS address, 4 rounds.
    for (int i = tid * 4; i < LL; i += 256 * 4) {
        const unsigned lds = (unsigned)(uintptr_t)(&vals[i]);
        const unsigned long long ga = (unsigned long long)(uintptr_t)(crow + i);
        asm volatile("global_load_async_to_lds_b128 %0, %1, off"
                     :: "v"(lds), "v"(ga) : "memory");
    }
    asm volatile("s_wait_asynccnt 0" ::: "memory");
    __syncthreads();

    for (int kk = 0; kk < TOPK; ++kk) {
        float bv = NEG_INF; int bi = 0;
        for (int i = tid; i < LL; i += 256) {
            const float vv = vals[i];
            if (vv > bv) { bv = vv; bi = i; }        // ascending i => lowest idx on tie
        }
        redv[tid] = bv; redi[tid] = bi;
        __syncthreads();
        for (int s = 128; s > 0; s >>= 1) {
            if (tid < s) {
                const float v2 = redv[tid + s]; const int i2 = redi[tid + s];
                if (v2 > redv[tid] || (v2 == redv[tid] && i2 < redi[tid])) {
                    redv[tid] = v2; redi[tid] = i2;
                }
            }
            __syncthreads();
        }
        if (tid == 0) { tv[kk] = redv[0]; ti[kk] = redi[0]; vals[redi[0]] = NEG_INF; }
        __syncthreads();
    }

    if (tid == 0) {
        float mx = tv[0];
        #pragma unroll
        for (int kk = 1; kk < TOPK; ++kk) mx = fmaxf(mx, tv[kk]);
        float e[TOPK]; float sum = 0.0f;
        #pragma unroll
        for (int kk = 0; kk < TOPK; ++kk) { e[kk] = __expf(tv[kk] - mx); sum += e[kk]; }
        const float inv = 1.0f / sum;
        #pragma unroll
        for (int kk = 0; kk < TOPK; ++kk) {
            wOut[(size_t)row * TOPK + kk] = e[kk] * inv;
            iOut[(size_t)row * TOPK + kk] = ti[kk];
        }
    }
}

// ---------------------------------------------------------------------------
// Kernel 4: weighted delay-gather accumulate.
// out[b,t,h*64+c] = sum_k w[b,h,c,k] * v[b, min(idx[b,h,c,k]+t, L-1), h*64+c]
// ---------------------------------------------------------------------------
__global__ __launch_bounds__(256)
void ac_aggregate(const float* __restrict__ v, const float* __restrict__ w,
                  const int* __restrict__ idx, float* __restrict__ out) {
    __shared__ float wl[DH][TOPK];
    __shared__ int   il[DH][TOPK];

    const int bh = blockIdx.y;                       // 0..63
    const int b  = bh >> 3, h = bh & 7;
    const int t0 = blockIdx.x * 256;
    const int tid = threadIdx.x;

    for (int i = tid; i < DH * TOPK; i += 256) {
        const int c = i >> 4, kk = i & 15;
        const int row = b * DD + h * DH + c;
        wl[c][kk] = w[(size_t)row * TOPK + kk];
        il[c][kk] = idx[(size_t)row * TOPK + kk];
    }
    __syncthreads();

    const int c    = tid & 63;
    const int tsub = tid >> 6;
    const size_t colbase = (size_t)h * DH + c;

    for (int tt = tsub; tt < 256; tt += 4) {
        const int t = t0 + tt;
        float acc = 0.0f;
        #pragma unroll
        for (int kk = 0; kk < TOPK; ++kk) {
            int g = il[c][kk] + t;
            if (g > LL - 1) g = LL - 1;              // saturating clamp
            acc += wl[c][kk] * v[((size_t)b * LL + g) * DD + colbase];
        }
        out[((size_t)b * LL + t) * DD + colbase] = acc;
    }
}

// ---------------------------------------------------------------------------
// Host launcher
// ---------------------------------------------------------------------------
extern "C" void kernel_launch(void* const* d_in, const int* in_sizes, int n_in,
                              void* d_out, int out_size, void* d_ws, size_t ws_size,
                              hipStream_t stream) {
    (void)in_sizes; (void)n_in; (void)out_size; (void)ws_size;

    const float* Q   = (const float*)d_in[0];
    const float* Kin = (const float*)d_in[1];
    const float* V   = (const float*)d_in[2];
    const float* Wq  = (const float*)d_in[3];
    const float* bq  = (const float*)d_in[4];
    const float* Wk  = (const float*)d_in[5];
    const float* bk  = (const float*)d_in[6];
    const float* Wv  = (const float*)d_in[7];
    const float* bv  = (const float*)d_in[8];

    const int M = BB * LL;                           // 32768
    const size_t nBLD = (size_t)M * DD;              // 16,777,216 floats (64 MB)
    const size_t nW   = (size_t)DD * DD;             // 262,144

    float* ws   = (float*)d_ws;
    float* qp   = ws;                                // [M, D]
    float* kp   = ws + nBLD;
    float* vp   = ws + 2 * nBLD;
    float* corr = ws + 3 * nBLD;                     // [B*D, L]
    float* wTop = ws + 4 * nBLD;                     // [B*D, 16]
    int*   iTop = (int*)(ws + 4 * nBLD + (size_t)BB * DD * TOPK);
    bf16_t* wqB = (bf16_t*)(ws + 4 * nBLD + 2 * (size_t)BB * DD * TOPK);
    bf16_t* wkB = wqB + nW;
    bf16_t* wvB = wkB + nW;

    // fp32 -> bf16 weight conversion (one-time per call)
    ac_w_convert<<<nW / 1024, 256, 0, stream>>>(Wq, wqB);
    ac_w_convert<<<nW / 1024, 256, 0, stream>>>(Wk, wkB);
    ac_w_convert<<<nW / 1024, 256, 0, stream>>>(Wv, wvB);

    const dim3 gemmGrid(DD / 64, M / 256);           // (8, 128)
    ac_proj_gemm<<<gemmGrid, 256, 0, stream>>>(Q,   wqB, bq, qp);
    ac_proj_gemm<<<gemmGrid, 256, 0, stream>>>(Kin, wkB, bk, kp);
    ac_proj_gemm<<<gemmGrid, 256, 0, stream>>>(V,   wvB, bv, vp);

    ac_fft_corr<<<BB * DD, 256, 0, stream>>>(qp, kp, corr);

    ac_topk_softmax<<<BB * DD, 256, 0, stream>>>(corr, wTop, iTop);

    ac_aggregate<<<dim3(LL / 256, BB * HH), 256, 0, stream>>>(vp, wTop, iTop, (float*)d_out);
}
